// PagedHeadAttention_11974368821410
// MI455X (gfx1250) — compile-verified, matrix operand order checked by reading, not run on hardware
//
#include <hip/hip_runtime.h>
#include <hip/hip_bf16.h>
#include <math.h>

// Problem constants (from reference)
#define BATCH   3
#define SEQ     4096
#define EMB     2048
#define HEAD    64
#define NROWS   (BATCH * SEQ)     // 12288 query rows
#define NTILES  (NROWS / 16)      // 768 row-tiles of 16

typedef __attribute__((ext_vector_type(16))) __bf16 v16bf;
typedef __attribute__((ext_vector_type(8)))  __bf16 v8bf;
typedef __attribute__((ext_vector_type(8)))  float  v8f;
typedef __attribute__((ext_vector_type(4)))  float  v4f;

// ---------------------------------------------------------------------------
// Kernel 1: split Wq (f32) into bf16 hi/lo planes (for bf16x3 WMMA GEMM)
// ---------------------------------------------------------------------------
__global__ void prep_w(const float* __restrict__ Wq,
                       __bf16* __restrict__ hi, __bf16* __restrict__ lo) {
  int i = blockIdx.x * blockDim.x + threadIdx.x;
  if (i < HEAD * EMB) {
    float w  = Wq[i];
    __bf16 h = (__bf16)w;
    hi[i] = h;
    lo[i] = (__bf16)(w - (float)h);
  }
}

// ---------------------------------------------------------------------------
// Kernel 2: k16/v16 = x[b, 0:16, :] @ Wk^T / Wv^T  (full f32 — only 16
// distinct K/V rows per batch survive the paged-cache semantics)
// Thread i -> (b, t, h) with i = b*1024 + t*64 + h
// ---------------------------------------------------------------------------
__global__ void prep_kv(const float* __restrict__ x,
                        const float* __restrict__ Wk,
                        const float* __restrict__ Wv,
                        float* __restrict__ k16, float* __restrict__ v16) {
  int i = blockIdx.x * blockDim.x + threadIdx.x;
  int h = i & 63;
  int t = (i >> 6) & 15;
  int b = i >> 10;
  if (b >= BATCH) return;
  const float* xr  = x  + ((size_t)b * SEQ + t) * EMB;
  const float* wkr = Wk + (size_t)h * EMB;
  const float* wvr = Wv + (size_t)h * EMB;
  float ak = 0.f, av = 0.f;
  for (int e = 0; e < EMB; e += 4) {
    v4f xv = *(const v4f*)(xr + e);
    v4f kv = *(const v4f*)(wkr + e);
    v4f vv = *(const v4f*)(wvr + e);
    ak += xv.x * kv.x + xv.y * kv.y + xv.z * kv.z + xv.w * kv.w;
    av += xv.x * vv.x + xv.y * vv.y + xv.z * vv.z + xv.w * vv.w;
  }
  k16[i] = ak;
  v16[i] = av;
}

// ---------------------------------------------------------------------------
// Kernel 3: fused Q-GEMM (WMMA bf16x3, f32 accum) + collapsed 16-key
// count-weighted causal softmax + V combine. One wave32 per 16-row tile.
// ---------------------------------------------------------------------------
__global__ __launch_bounds__(256) void attn_main(
    const float*  __restrict__ x,
    const __bf16* __restrict__ wq_hi,
    const __bf16* __restrict__ wq_lo,
    const float*  __restrict__ k16,
    const float*  __restrict__ v16,
    float*        __restrict__ out) {
  __shared__ float qtile[8 * 16 * 65];          // 8 waves, 16x64 padded to 65

  const int wave = threadIdx.x >> 5;
  const int lane = threadIdx.x & 31;
  const int r    = lane & 15;                   // row within tile / N index
  const int kh   = lane >> 4;                   // lane-half: selects K sub-blocks
  const int tile = blockIdx.x * 8 + wave;       // 0..767, exact
  const int m0   = tile * 16;

  const float* xrow = x + (size_t)(m0 + r) * EMB;

  v8f acc[4] = {};                              // 16x64 f32 Q accumulators

  for (int kb = 0; kb < EMB; kb += 32) {
    // ---- A operand: x row, ISA 16-bit A layout.
    // lanes 0-15: K = {0..7, 16..23}; lanes 16-31: K = {8..15, 24..31}
    const float* ap = xrow + kb + kh * 8;
    float af[16];
    *(v4f*)(af + 0)  = *(const v4f*)(ap + 0);
    *(v4f*)(af + 4)  = *(const v4f*)(ap + 4);
    *(v4f*)(af + 8)  = *(const v4f*)(ap + 16);
    *(v4f*)(af + 12) = *(const v4f*)(ap + 20);
    v16bf ah, al;
#pragma unroll
    for (int i = 0; i < 16; i++) {
      float  f = af[i];
      __bf16 h = (__bf16)f;
      ah[i] = h;
      al[i] = (__bf16)(f - (float)h);           // exact residual, then bf16
    }

    // ---- 4 N-tiles of 16 head columns each; B column n == Wq row n (contig)
#pragma unroll
    for (int j = 0; j < 4; j++) {
      const size_t  boff = (size_t)(j * 16 + r) * EMB + kb + kh * 8;
      const __bf16* bhp  = wq_hi + boff;
      const __bf16* blp  = wq_lo + boff;
      v8bf bh0 = *(const v8bf*)(bhp);
      v8bf bh1 = *(const v8bf*)(bhp + 16);
      v8bf bl0 = *(const v8bf*)(blp);
      v8bf bl1 = *(const v8bf*)(blp + 16);
      v16bf bhi = __builtin_shufflevector(bh0, bh1, 0,1,2,3,4,5,6,7,8,9,10,11,12,13,14,15);
      v16bf blo = __builtin_shufflevector(bl0, bl1, 0,1,2,3,4,5,6,7,8,9,10,11,12,13,14,15);

      // bf16x3: xh*wh + xl*wh + xh*wl  (f32 accumulate)
      acc[j] = __builtin_amdgcn_wmma_f32_16x16x32_bf16(false, ah, false, bhi,
                                                       (short)0, acc[j], false, false);
      acc[j] = __builtin_amdgcn_wmma_f32_16x16x32_bf16(false, al, false, bhi,
                                                       (short)0, acc[j], false, false);
      acc[j] = __builtin_amdgcn_wmma_f32_16x16x32_bf16(false, ah, false, blo,
                                                       (short)0, acc[j], false, false);
    }
  }

  // ---- Spill Q tile to LDS. D layout: VGPR i -> M = i + 8*kh, N = j*16 + r
  float* qt = qtile + wave * 16 * 65;
#pragma unroll
  for (int j = 0; j < 4; j++) {
#pragma unroll
    for (int i = 0; i < 8; i++) {
      qt[(kh * 8 + i) * 65 + j * 16 + r] = acc[j][i];
    }
  }
  __syncthreads();

  // ---- Collapsed attention: 16 distinct keys, count-weighted softmax.
  // Lane handles row r; kh selects output halves h in [kh*32, kh*32+32).
  const int m  = m0 + r;
  const int b  = m >> 12;                       // / SEQ (4096)
  const int qi = m & (SEQ - 1);

  float qreg[64];
  const float* qv = qt + r * 65;
#pragma unroll
  for (int h = 0; h < 64; h++) qreg[h] = qv[h];

  const float* kb16 = k16 + b * 16 * 64;
  const float* vb16 = v16 + b * 16 * 64;
  const int d    = qi >> 4;
  const int mres = qi & 15;

  float s[16];
  float smax = -3.0e38f;
#pragma unroll
  for (int t = 0; t < 16; t++) {
    float dot = 0.f;
    const float* kr = kb16 + t * 64;
    for (int h = 0; h < 64; h++) dot += qreg[h] * kr[h];
    s[t] = dot * 0.125f;                        // HEAD^-0.5 = 1/8
    int cnt = d + ((t <= mres) ? 1 : 0);
    if (cnt > 0 && s[t] > smax) smax = s[t];
  }

  float w[16];
  float Z = 0.f;
#pragma unroll
  for (int t = 0; t < 16; t++) {
    int cnt = d + ((t <= mres) ? 1 : 0);
    float e = 0.f;
    if (cnt > 0) e = (float)cnt * expf(s[t] - smax);
    w[t] = e;
    Z += e;
  }
  const float invZ = 1.0f / Z;

  float* orow = out + (size_t)m * HEAD + kh * 32;
  for (int h = 0; h < 32; h++) {
    float o = 0.f;
#pragma unroll
    for (int t = 0; t < 16; t++) o += w[t] * vb16[t * 64 + kh * 32 + h];
    orow[h] = o * invZ;
  }
}

// ---------------------------------------------------------------------------
extern "C" void kernel_launch(void* const* d_in, const int* in_sizes, int n_in,
                              void* d_out, int out_size, void* d_ws, size_t ws_size,
                              hipStream_t stream) {
  const float* x  = (const float*)d_in[0];
  const float* Wq = (const float*)d_in[1];
  const float* Wk = (const float*)d_in[2];
  const float* Wv = (const float*)d_in[3];
  float* out = (float*)d_out;

  char* ws = (char*)d_ws;
  // ws layout: wq_hi (256KB) | wq_lo (256KB) | k16 (12KB) | v16 (12KB)
  __bf16* wq_hi = (__bf16*)(ws);
  __bf16* wq_lo = (__bf16*)(ws + 262144);
  float*  k16   = (float*) (ws + 524288);
  float*  v16   = (float*) (ws + 524288 + 12288);

  prep_w   <<<(HEAD * EMB + 255) / 256, 256, 0, stream>>>(Wq, wq_hi, wq_lo);
  prep_kv  <<<(BATCH * 16 * HEAD + 255) / 256, 256, 0, stream>>>(x, Wk, Wv, k16, v16);
  attn_main<<<NTILES / 8, 256, 0, stream>>>(x, wq_hi, wq_lo, k16, v16, out);
}
